// LlamaAttention_4561255268357
// MI455X (gfx1250) — matrix operand order repaired and checked by reading, not compiled, not run on hardware
//
#include <hip/hip_runtime.h>
#include <math.h>

typedef __bf16 bf16;
typedef __attribute__((ext_vector_type(16))) __bf16 bf16x16;
typedef __attribute__((ext_vector_type(8)))  __bf16 bf16x8;
typedef __attribute__((ext_vector_type(8)))  float  v8f;

#define HIDDEN 2048
#define NHEADS 32
#define NKV    8
#define HD     64
#define SEQ    2048
#define BATCH  2
#define ROWS   (BATCH*SEQ)   // 4096

// ---------- helpers ----------

static __device__ __forceinline__ bf16 f2bf(float f) {
  unsigned u = __builtin_bit_cast(unsigned, f);
  u += 0x7FFFu + ((u >> 16) & 1u);            // round-to-nearest-even
  unsigned short h = (unsigned short)(u >> 16);
  return __builtin_bit_cast(bf16, h);
}

static __device__ __forceinline__ v8f wmma_bf16(bf16x16 a, bf16x16 b, v8f c) {
  // D(16x16,f32) = A(16x32,bf16) * B(32x16,bf16) + C
  return __builtin_amdgcn_wmma_f32_16x16x32_bf16(false, a, false, b, (short)0, c,
                                                 false, false);
}

static __device__ __forceinline__ bf16x16 cat8(bf16x8 lo, bf16x8 hi) {
  return __builtin_shufflevector(lo, hi, 0,1,2,3,4,5,6,7,8,9,10,11,12,13,14,15);
}
static __device__ __forceinline__ bf16x8  ld8 (const bf16* p){ return *(const bf16x8 *)p; }
static __device__ __forceinline__ bf16x16 ld16(const bf16* p){ return *(const bf16x16*)p; }

// CDNA5 async global->LDS DMA (ASYNCcnt-tracked, ISA §15.18.3).
// The LDS *pointer* is passed (ptrtoint escape) so the compiler cannot fold the
// staged buffers to undef; "memory" clobber orders the later ds_loads after it.
static __device__ __forceinline__ void async_ld_b128(void* ldsp, const void* g) {
  asm volatile("global_load_async_to_lds_b128 %0, %1, off"
               :: "v"((unsigned)(size_t)ldsp), "v"(g)
               : "memory");
}
// Async loads complete in order per wave: waiting ASYNCcnt<=N leaves the newest
// N DMAs in flight while guaranteeing the older ones have landed in LDS.
static __device__ __forceinline__ void wait_async_le0() {
  asm volatile("s_wait_asynccnt 0x0" ::: "memory");
}
static __device__ __forceinline__ void wait_async_le2() {
  asm volatile("s_wait_asynccnt 0x2" ::: "memory");
}
static __device__ __forceinline__ void wait_async_le4() {
  asm volatile("s_wait_asynccnt 0x4" ::: "memory");
}

// ---------- cooperative GEMM tile ----------
// 4-wave workgroup. Each wave computes 16 rows x 64 cols; the 64-col x 32-k B
// chunk (4KB) is staged once per workgroup into LDS via async DMA (double-
// buffered: next chunk's DMA overlaps current chunk's WMMAs) and shared.
// LDS buf layout: [64 cols][32 k] bf16 (64B per col).
static __device__ __forceinline__ void stage_b(const bf16* BT, int ldb, int kc,
                                               bf16* buf, int w, int lane) {
#pragma unroll
  for (int i = 0; i < 2; ++i) {
    int col = (w * 2 + i) * 8 + (lane >> 2);          // 0..63
    unsigned bo = (unsigned)((lane & 3) * 16);
    async_ld_b128((char*)buf + col * 64 + bo,
                  (const char*)(BT + (size_t)col * ldb + kc) + bo);
  }
}

static __device__ __forceinline__ void gemm16x64_coop(
    const bf16* __restrict__ A, int lda,
    const bf16* __restrict__ BT, int ldb,
    int ktot, int tid, bf16* bbuf /* 2*64*32 bf16 */, v8f acc[4])
{
  const int lane = tid & 31, w = tid >> 5;
  const int n = lane & 15, g = lane >> 4;
  const bf16* arow = A + (size_t)n * lda;
  v8f z = {};
  acc[0] = z; acc[1] = z; acc[2] = z; acc[3] = z;

  stage_b(BT, ldb, 0, bbuf, w, lane);
  int cur = 0;
  for (int kc = 0; kc < ktot; kc += 32) {
    if (kc + 32 < ktot) {
      stage_b(BT, ldb, kc + 32, bbuf + (cur ^ 1) * (64 * 32), w, lane); // next DMA
      wait_async_le2();                                 // current chunk landed
      __builtin_prefetch(arow + kc + 32, 0, 3);
    } else {
      wait_async_le0();
    }
    __syncthreads();                     // all waves see staged chunk
    const bf16* bc = bbuf + cur * (64 * 32);
    // A-frag: k(e) = g*16 + e
    bf16x16 afrag = ld16(arow + kc + g * 16);
#pragma unroll
    for (int t = 0; t < 4; ++t) {
      // B-frag from LDS: k(e) = g*16 + e, contiguous 32B
      bf16x16 bfrag = ld16(bc + (size_t)(t * 16 + n) * 32 + g * 16);
      acc[t] = wmma_bf16(afrag, bfrag, acc[t]);
    }
    __syncthreads();                     // done reading before this buf restages
    cur ^= 1;
  }
}

// ---------- kernel 1: fp32 -> bf16 convert / transpose ----------

__global__ void cvt_bf16_kernel(const float* __restrict__ in, bf16* __restrict__ out, int nelem) {
  int i = blockIdx.x * blockDim.x + threadIdx.x;
  if (i < nelem) out[i] = f2bf(in[i]);
}

__global__ void transpose_bf16_kernel(const float* __restrict__ in, bf16* __restrict__ out,
                                      int K, int N) {
  int i = blockIdx.x * blockDim.x + threadIdx.x;
  if (i < K * N) {
    int k = i / N, n = i - k * N;
    out[(size_t)n * K + k] = f2bf(in[i]);   // out is N-major [N][K]
  }
}

// ---------- kernel 2: fused QKV projection + RoPE ----------
// grid: (ROWS/64, 32 Q-heads + 8 K-heads + 8 V-heads), block: 128 (4 waves)

__global__ void __launch_bounds__(128) qkv_rope_kernel(
    const bf16* __restrict__ X,   const bf16* __restrict__ wqT,
    const bf16* __restrict__ wkT, const bf16* __restrict__ wvT,
    bf16* __restrict__ Q, bf16* __restrict__ K, bf16* __restrict__ VT)
{
  __shared__ __align__(64) bf16 bbuf[2 * 64 * 32];
  const int tid = threadIdx.x;
  const int lane = tid & 31, w = tid >> 5;
  const int n = lane & 15, g = lane >> 4;
  const int rowGrp = blockIdx.x;            // 64-row group
  const int slot = blockIdx.y;

  const bf16* BT; int head, mode;
  if (slot < NHEADS)            { BT = wqT; head = slot;          mode = 0; }
  else if (slot < NHEADS + NKV) { BT = wkT; head = slot - NHEADS; mode = 1; }
  else                          { BT = wvT; head = slot - NHEADS - NKV; mode = 2; }

  v8f acc[4];
  gemm16x64_coop(X + ((size_t)rowGrp * 64 + w * 16) * HIDDEN, HIDDEN,
                 BT + (size_t)head * HD * HIDDEN, HIDDEN, HIDDEN, tid, bbuf, acc);

  float res[4][8];
  if (mode == 2) {
#pragma unroll
    for (int t = 0; t < 4; ++t)
#pragma unroll
      for (int r = 0; r < 8; ++r) res[t][r] = acc[t][r];
  } else {
    // RoPE: d<32 pairs with d+32 (tile t <-> t^2)
#pragma unroll
    for (int t = 0; t < 4; ++t) {
      const int dh = t * 16 + n;                       // 0..63 within head
      const float invf = __expf((float)(dh & 31) * (-9.210340371976184f / 32.0f));
      const float sgn  = (dh < 32) ? -1.0f : 1.0f;
#pragma unroll
      for (int r = 0; r < 8; ++r) {
        int row = rowGrp * 64 + w * 16 + r + g * 8;
        float pos = (float)(row & (SEQ - 1));
        float ang = pos * invf;
        res[t][r] = acc[t][r] * __cosf(ang) + sgn * acc[t ^ 2][r] * __sinf(ang);
      }
    }
  }

#pragma unroll
  for (int t = 0; t < 4; ++t) {
    const int dh = t * 16 + n;
#pragma unroll
    for (int r = 0; r < 8; ++r) {
      int row = rowGrp * 64 + w * 16 + r + g * 8;
      int b = row >> 11, s = row & (SEQ - 1);
      bf16 v = f2bf(res[t][r]);
      if (mode == 0)      Q [(((size_t)b * NHEADS + head) * SEQ + s) * HD + dh] = v;
      else if (mode == 1) K [(((size_t)b * NKV    + head) * SEQ + s) * HD + dh] = v;
      else                VT[(((size_t)b * NKV    + head) * HD  + dh) * SEQ + s] = v; // transposed
    }
  }
}

// ---------- kernel 3: flash attention ----------
// grid: (SEQ/64, NHEADS, BATCH), block: 128 (4 waves; wave w owns q-tile 4*bx+w).
// Double-buffered KV chunks staged per workgroup via async DMA, shared by all
// 4 q-tiles; DMA of chunk i+1 overlaps softmax+WMMA of chunk i.

__global__ void __launch_bounds__(128) attn_kernel(
    const bf16* __restrict__ Q, const bf16* __restrict__ K,
    const bf16* __restrict__ VT, bf16* __restrict__ AO)
{
  __shared__ __align__(64) bf16 smem[2 * 32 * 64 + 2 * 64 * 32 + 4 * 16 * 32];
  // layout: K ping/pong [32 kv][64 d] | V ping/pong [64 d][32 kv] | P[4 waves]
  const int tid = threadIdx.x, lane = tid & 31, w = tid >> 5;
  const int n = lane & 15, g = lane >> 4;
  bf16* pbuf = smem + 2 * 32 * 64 + 2 * 64 * 32 + w * (16 * 32); // per-wave P
  const int qt = blockIdx.x * 4 + w, h = blockIdx.y, b = blockIdx.z;

  const bf16* Qh = Q  + (((size_t)b * NHEADS + h)        * SEQ) * HD;
  const bf16* Kh = K  + (((size_t)b * NKV    + (h >> 2)) * SEQ) * HD;
  const bf16* Vh = VT + (((size_t)b * NKV    + (h >> 2)) * HD ) * SEQ;

  // Q A-fragments, resident for the whole loop
  const bf16* qrow = Qh + (size_t)(qt * 16 + n) * HD;
  bf16x16 qa0 = ld16(qrow +      g * 16);
  bf16x16 qa1 = ld16(qrow + 32 + g * 16);

  v8f z = {};
  v8f o0 = z, o1 = z, o2 = z, o3 = z;
  float mrow[8], lrow[8];
#pragma unroll
  for (int r = 0; r < 8; ++r) { mrow[r] = -1e30f; lrow[r] = 0.0f; }

  // KV chunk staging: K = 4KB contiguous, V = 64 rows (d) x 64B strided SEQ
  auto stage_kv = [&](int kc, int buf) {
    bf16* kb = smem + buf * (32 * 64);
    bf16* vb = smem + 2 * 32 * 64 + buf * (64 * 32);
#pragma unroll
    for (int i = 0; i < 2; ++i) {
      unsigned off = (unsigned)(((w * 2 + i) * 512) + lane * 16);
      async_ld_b128((char*)kb + off, (const char*)(Kh + (size_t)kc * HD) + off);
    }
#pragma unroll
    for (int i = 0; i < 2; ++i) {
      int row = (w * 2 + i) * 8 + (lane >> 2);        // 0..63
      unsigned bo = (unsigned)((lane & 3) * 16);
      async_ld_b128((char*)vb + row * 64 + bo,
                    (const char*)(Vh + (size_t)row * SEQ + kc) + bo);
    }
  };

  stage_kv(0, 0);
  int cur = 0;
  for (int kc = 0; kc < SEQ; kc += 32) {
    if (kc + 32 < SEQ) {
      stage_kv(kc + 32, cur ^ 1);        // overlap next chunk's DMA
      wait_async_le4();                  // current 4 landed
    } else {
      wait_async_le0();
    }
    __syncthreads();
    const bf16* kbuf = smem + cur * (32 * 64);
    const bf16* vbuf = smem + 2 * 32 * 64 + cur * (64 * 32);

    // K^T B-fragments from LDS (two 16-kv tiles x two 32-d chunks)
    bf16x16 kb00 = ld16(kbuf + (size_t)(     n) * HD +      g * 16);
    bf16x16 kb01 = ld16(kbuf + (size_t)(     n) * HD + 32 + g * 16);
    bf16x16 kb10 = ld16(kbuf + (size_t)(16 + n) * HD +      g * 16);
    bf16x16 kb11 = ld16(kbuf + (size_t)(16 + n) * HD + 32 + g * 16);

    v8f s0 = wmma_bf16(qa0, kb00, z); s0 = wmma_bf16(qa1, kb01, s0);
    v8f s1 = wmma_bf16(qa0, kb10, z); s1 = wmma_bf16(qa1, kb11, s1);

    // online softmax; lane handles rows r+g*8, reduce across the 16-lane half
#pragma unroll
    for (int r = 0; r < 8; ++r) {
      float a0 = s0[r] * 0.125f, a1 = s1[r] * 0.125f;   // 1/sqrt(64)
      float mx = fmaxf(a0, a1);
      mx = fmaxf(mx, __shfl_xor(mx, 1));
      mx = fmaxf(mx, __shfl_xor(mx, 2));
      mx = fmaxf(mx, __shfl_xor(mx, 4));
      mx = fmaxf(mx, __shfl_xor(mx, 8));
      float mnew  = fmaxf(mrow[r], mx);
      float alpha = __expf(mrow[r] - mnew);
      float p0 = __expf(a0 - mnew), p1 = __expf(a1 - mnew);
      float rs = p0 + p1;
      rs += __shfl_xor(rs, 1);
      rs += __shfl_xor(rs, 2);
      rs += __shfl_xor(rs, 4);
      rs += __shfl_xor(rs, 8);
      lrow[r] = lrow[r] * alpha + rs;
      mrow[r] = mnew;
      o0[r] *= alpha; o1[r] *= alpha; o2[r] *= alpha; o3[r] *= alpha;
      int m = r + g * 8;
      pbuf[m * 32 +      n] = f2bf(p0);
      pbuf[m * 32 + 16 + n] = f2bf(p1);
    }
    // reload P as A-fragment (same-wave LDS ops are in order)
    bf16x16 pa = ld16(pbuf + n * 32 + g * 16);

    // PV from staged transposed V
    o0 = wmma_bf16(pa, ld16(vbuf + (size_t)( 0 + n) * 32 + g * 16), o0);
    o1 = wmma_bf16(pa, ld16(vbuf + (size_t)(16 + n) * 32 + g * 16), o1);
    o2 = wmma_bf16(pa, ld16(vbuf + (size_t)(32 + n) * 32 + g * 16), o2);
    o3 = wmma_bf16(pa, ld16(vbuf + (size_t)(48 + n) * 32 + g * 16), o3);

    __syncthreads();           // all waves done with this buffer before restage
    cur ^= 1;
  }

#pragma unroll
  for (int r = 0; r < 8; ++r) {
    float inv = 1.0f / lrow[r];
    size_t row  = (size_t)b * SEQ + qt * 16 + r + g * 8;
    size_t base = row * HIDDEN + (size_t)h * HD + n;
    AO[base +  0] = f2bf(o0[r] * inv);
    AO[base + 16] = f2bf(o1[r] * inv);
    AO[base + 32] = f2bf(o2[r] * inv);
    AO[base + 48] = f2bf(o3[r] * inv);
  }
}

// ---------- kernel 4: output projection ----------
// grid: (ROWS/64, HIDDEN/64), block: 128 (4 waves)

__global__ void __launch_bounds__(128) proj_kernel(
    const bf16* __restrict__ A, const bf16* __restrict__ woT, float* __restrict__ out)
{
  __shared__ __align__(64) bf16 bbuf[2 * 64 * 32];
  const int tid = threadIdx.x;
  const int lane = tid & 31, w = tid >> 5;
  const int n = lane & 15, g = lane >> 4;
  const int rowGrp = blockIdx.x, ngrp = blockIdx.y;
  v8f acc[4];
  gemm16x64_coop(A   + ((size_t)rowGrp * 64 + w * 16) * HIDDEN, HIDDEN,
                 woT + (size_t)ngrp * HD * HIDDEN, HIDDEN, HIDDEN, tid, bbuf, acc);
#pragma unroll
  for (int t = 0; t < 4; ++t)
#pragma unroll
    for (int r = 0; r < 8; ++r) {
      size_t row = (size_t)rowGrp * 64 + w * 16 + r + g * 8;
      out[row * HIDDEN + ngrp * HD + t * 16 + n] = acc[t][r];
    }
}

// ---------- launch ----------

extern "C" void kernel_launch(void* const* d_in, const int* in_sizes, int n_in,
                              void* d_out, int out_size, void* d_ws, size_t ws_size,
                              hipStream_t stream) {
  (void)in_sizes; (void)n_in; (void)out_size; (void)ws_size;
  const float* x  = (const float*)d_in[0];
  const float* wq = (const float*)d_in[1];
  const float* wk = (const float*)d_in[2];
  const float* wv = (const float*)d_in[3];
  const float* wo = (const float*)d_in[4];
  float* out = (float*)d_out;

  char* ws = (char*)d_ws;
  size_t off = 0;
  auto carve = [&](size_t bytes) -> void* {
    void* p = ws + off;
    off += (bytes + 255) & ~(size_t)255;
    return p;
  };
  bf16* Xb  = (bf16*)carve((size_t)ROWS * HIDDEN * 2);          // 16 MB
  bf16* wqT = (bf16*)carve((size_t)HIDDEN * HIDDEN * 2);        //  8 MB
  bf16* wkT = (bf16*)carve((size_t)HIDDEN * NKV * HD * 2);      //  2 MB
  bf16* wvT = (bf16*)carve((size_t)HIDDEN * NKV * HD * 2);      //  2 MB
  bf16* woT = (bf16*)carve((size_t)HIDDEN * HIDDEN * 2);        //  8 MB
  bf16* Qw  = (bf16*)carve((size_t)ROWS * HIDDEN * 2);          // 16 MB
  bf16* Kw  = (bf16*)carve((size_t)BATCH * NKV * SEQ * HD * 2); //  4 MB
  bf16* VTw = (bf16*)carve((size_t)BATCH * NKV * SEQ * HD * 2); //  4 MB
  bf16* AOw = (bf16*)carve((size_t)ROWS * HIDDEN * 2);          // 16 MB

  const int nx = ROWS * HIDDEN;
  cvt_bf16_kernel<<<(nx + 255) / 256, 256, 0, stream>>>(x, Xb, nx);
  transpose_bf16_kernel<<<(HIDDEN * HIDDEN + 255) / 256, 256, 0, stream>>>(wq, wqT, HIDDEN, HIDDEN);
  transpose_bf16_kernel<<<(HIDDEN * NKV * HD + 255) / 256, 256, 0, stream>>>(wk, wkT, HIDDEN, NKV * HD);
  transpose_bf16_kernel<<<(HIDDEN * NKV * HD + 255) / 256, 256, 0, stream>>>(wv, wvT, HIDDEN, NKV * HD);
  transpose_bf16_kernel<<<(HIDDEN * HIDDEN + 255) / 256, 256, 0, stream>>>(wo, woT, HIDDEN, HIDDEN);

  qkv_rope_kernel<<<dim3(ROWS / 64, NHEADS + 2 * NKV), 128, 0, stream>>>(
      Xb, wqT, wkT, wvT, Qw, Kw, VTw);
  attn_kernel<<<dim3(SEQ / 64, NHEADS, BATCH), 128, 0, stream>>>(Qw, Kw, VTw, AOw);
  proj_kernel<<<dim3(ROWS / 64, HIDDEN / 64), 128, 0, stream>>>(AOw, woT, out);
}
